// MuleHunterGNN_50079318671882
// MI455X (gfx1250) — matrix operand order repaired and checked by reading, not compiled, not run on hardware
//
#include <hip/hip_runtime.h>
#include <hip/hip_bf16.h>
#include <math.h>

// ---------------------------------------------------------------------------
// MuleHunter GNN for gfx1250 (MI455X, wave32, WMMA).
//  - dense projections: v_wmma_f32_16x16x32_f16, weights pre-packed to f16
//    B-fragment layout once per launch (L2-resident, 2x b128 per fragment)
//  - h1 stored directly as f16 A-fragments (only consumer is the GAT GEMM)
//  - segment ops: atomics; L2 warmed with global_prefetch_b8
// ---------------------------------------------------------------------------

#define GN 50000
#define GE 800000
#define GIN 20
#define GHID 64
#define GHEADS 4
#define GHD 64
#define GPROJ (GHEADS * GHD)   // 256
#define GH3 (GHID / 2)         // 32
#define GOUT 2
#define GEPS 1e-5f

typedef __attribute__((ext_vector_type(16))) _Float16 v16h;
typedef __attribute__((ext_vector_type(8)))  float    v8f;

// ---------------- WMMA operand layouts (ISA 7.12.2) -------------------------
// A (16x32 f16): lane&15 = row M; element i holds
//   K = (i<8 ? i : i+8) + (lane<16 ? 0 : 8)
// B (32x16 f16): lane&15 = col N; element i holds K = (i<8 ? i : i+8) + (lane<16 ? 0 : 8), same as A
// C/D (16x16 f32): row = r + 8*(lane>=16), col = lane&15

__device__ __forceinline__ int a_kidx(int i, int lane) {
    return (i < 8 ? i : i + 8) + ((lane < 16) ? 0 : 8);
}

// build A fragment from row-major f32 (for atomics-built buffers)
__device__ __forceinline__ v16h load_a_row(const float* rowbase, int kOff, int kLimit,
                                           int lane, float scale) {
    v16h a;
#pragma unroll
    for (int i = 0; i < 16; ++i) {
        int k = a_kidx(i, lane) + kOff;
        float v = (k < kLimit) ? rowbase[k] * scale : 0.0f;
        a[i] = (_Float16)v;
    }
    return a;
}

#define WMMA_F16(A, B, C) \
    __builtin_amdgcn_wmma_f32_16x16x32_f16(false, (A), false, (B), (short)0, (C), false, false)

// fragment fetch from pre-packed f16 buffers: frag block = 32 lanes x 32B
__device__ __forceinline__ v16h load_frag(const _Float16* pack, int frag, int lane) {
    return ((const v16h*)pack)[frag * 32 + lane];
}

// ---------------------------- utility kernels -------------------------------

__global__ void k_fill(float* __restrict__ p, long n, float v) {
    long i = (long)blockIdx.x * blockDim.x + threadIdx.x;
    if (i < n) p[i] = v;
}

// pack f32 row-major weight W[KIN x NOUT] into f16 B-fragment layout:
// frag = t*KCH + c ; lane entry = 16 halves (32B)
__global__ void k_packB(const float* __restrict__ W, int KIN, int NOUT,
                        _Float16* __restrict__ out, int nfrag, int KCH) {
    int idx = blockIdx.x * blockDim.x + threadIdx.x;
    if (idx >= nfrag * 32) return;
    int frag = idx >> 5, lane = idx & 31;
    int t = frag / KCH, c = frag % KCH;
    int n = t * 16 + (lane & 15);
    v16h b;
#pragma unroll
    for (int i = 0; i < 16; ++i) {
        int k = (i < 8 ? i : i + 8) + ((lane < 16) ? 0 : 8) + c * 32;
        float v = (k < KIN) ? W[k * NOUT + n] : 0.0f;
        b[i] = (_Float16)v;
    }
    ((v16h*)out)[idx] = b;
}

__global__ void k_degree(const int* __restrict__ dst, const int* __restrict__ src,
                         const float* __restrict__ x, float* __restrict__ deg, int E_) {
    int e = blockIdx.x * blockDim.x + threadIdx.x;
    if (e >= E_) return;
    atomicAdd(&deg[dst[e]], 1.0f);
    // warm L2 with the x[src] rows the next kernel gathers (global_prefetch_b8)
    __builtin_prefetch(x + (long)src[e] * GIN, 0, 1);
}

// scatter-add of F-wide features along edges: out[dst] += feat[src]
__global__ void k_scatter_add(const float* __restrict__ feat,
                              const int* __restrict__ src, const int* __restrict__ dst,
                              float* __restrict__ out, long total, int F) {
    long i = (long)blockIdx.x * blockDim.x + threadIdx.x;
    if (i >= total) return;
    int e = (int)(i / F);
    int c = (int)(i % F);
    atomicAdd(&out[(long)dst[e] * F + c], feat[(long)src[e] * F + c]);
}

__device__ __forceinline__ void atomicMaxFloat(float* addr, float v) {
    unsigned int* ua = (unsigned int*)addr;
    unsigned int old = *ua;
    for (;;) {
        float f = __uint_as_float(old);
        if (f >= v) break;
        unsigned int assumed = old;
        old = atomicCAS(ua, assumed, __float_as_uint(v));
        if (old == assumed) break;
    }
}

// ------------------------- SAGE conv (WMMA GEMM) ----------------------------
// out = relu(BN((agg/deg)@Wl + root@Wr + bias)); one 16-row tile per wave32.
// PACKOUT: store result as f16 A-fragments (consumed only by the next GEMM).
template <int KIN, int NOUT, bool PACKOUT>
__global__ void k_sage(const float* __restrict__ agg, const float* __restrict__ root,
                       const float* __restrict__ deg,
                       const _Float16* __restrict__ Wlp, const _Float16* __restrict__ Wrp,
                       const float* __restrict__ bias, const float* __restrict__ gamma,
                       const float* __restrict__ beta,
                       float* __restrict__ outF, _Float16* __restrict__ outP) {
    constexpr int KCH  = (KIN + 31) / 32;
    constexpr int NT   = NOUT / 16;
    constexpr int KCHO = NOUT / 32;
    int lane = threadIdx.x;
    int m0   = blockIdx.x * 16;
    int rowA = m0 + (lane & 15);
    float invdeg = 1.0f / fmaxf(deg[rowA], 1.0f);

    v16h Aagg[KCH], Aroot[KCH];
#pragma unroll
    for (int c = 0; c < KCH; ++c) {
        Aagg[c]  = load_a_row(agg  + (long)rowA * KIN, c * 32, KIN, lane, invdeg);
        Aroot[c] = load_a_row(root + (long)rowA * KIN, c * 32, KIN, lane, 1.0f);
    }
#pragma unroll
    for (int t = 0; t < NT; ++t) {
        v8f acc = {};
#pragma unroll
        for (int c = 0; c < KCH; ++c) {
            acc = WMMA_F16(Aagg[c],  load_frag(Wlp, t * KCH + c, lane), acc);
            acc = WMMA_F16(Aroot[c], load_frag(Wrp, t * KCH + c, lane), acc);
        }
        int n = t * 16 + (lane & 15);
        float scale = gamma[n] * rsqrtf(1.0f + GEPS);
#pragma unroll
        for (int r = 0; r < 8; ++r) {
            int m_local = r + ((lane >> 4) << 3);
            float v = fmaxf((acc[r] + bias[n]) * scale + beta[n], 0.0f);
            if (PACKOUT) {
                // scatter into A-fragment layout of the next GEMM
                int c2 = n >> 5, kk = n & 31;
                int hi = (kk >> 3) & 1;
                int kkp = kk - (hi << 3);
                int i2 = (kkp < 8) ? kkp : kkp - 8;
                int lane_p = m_local + (hi << 4);
                outP[(((long)blockIdx.x * KCHO + c2) * 32 + lane_p) * 16 + i2] = (_Float16)v;
            } else {
                outF[(long)(m0 + m_local) * NOUT + n] = v;
            }
        }
    }
}

// ------------------------- GAT projection (WMMA) ----------------------------
// hp = h1 @ W2 (50000x64 @ 64x256); A from pre-packed f16 fragments.
__global__ void k_gatproj(const _Float16* __restrict__ h1p, const _Float16* __restrict__ W2p,
                          float* __restrict__ hp) {
    int lane = threadIdx.x;
    int m0   = blockIdx.x * 16;
    v16h A0 = load_frag(h1p, blockIdx.x * 2 + 0, lane);
    v16h A1 = load_frag(h1p, blockIdx.x * 2 + 1, lane);
#pragma unroll
    for (int t = 0; t < 4; ++t) {
        int nt = blockIdx.y * 4 + t;        // global n-tile, 0..15
        v8f acc = {};
        acc = WMMA_F16(A0, load_frag(W2p, nt * 2 + 0, lane), acc);
        acc = WMMA_F16(A1, load_frag(W2p, nt * 2 + 1, lane), acc);
        int n = nt * 16 + (lane & 15);
#pragma unroll
        for (int r = 0; r < 8; ++r) {
            int m = m0 + r + ((lane >> 4) << 3);
            hp[(long)m * GPROJ + n] = acc[r];
        }
    }
}

// per-(node,head) attention logits
__global__ void k_attdot(const float* __restrict__ hp, const float* __restrict__ att_src,
                         const float* __restrict__ att_dst, float* __restrict__ a_s,
                         float* __restrict__ a_d, int total) {
    int i = blockIdx.x * blockDim.x + threadIdx.x;
    if (i >= total) return;
    int n = i >> 2, h = i & 3;
    const float* p  = hp + (long)n * GPROJ + h * GHD;
    const float* as = att_src + h * GHD;
    const float* ad = att_dst + h * GHD;
    float s = 0.0f, d = 0.0f;
    for (int k = 0; k < GHD; ++k) { s += p[k] * as[k]; d += p[k] * ad[k]; }
    a_s[i] = s; a_d[i] = d;
}

__device__ __forceinline__ float leaky(float v) { return v > 0.0f ? v : 0.2f * v; }

__global__ void k_edge_max(const float* __restrict__ a_s, const float* __restrict__ a_d,
                           const int* __restrict__ src, const int* __restrict__ dst,
                           float* __restrict__ mmax, long total) {
    long i = (long)blockIdx.x * blockDim.x + threadIdx.x;
    if (i >= total) return;
    int e = (int)(i >> 2), h = (int)(i & 3);
    float v = leaky(a_s[(long)src[e] * 4 + h] + a_d[(long)dst[e] * 4 + h]);
    atomicMaxFloat(&mmax[(long)dst[e] * 4 + h], v);
}

__global__ void k_edge_exp(const float* __restrict__ a_s, const float* __restrict__ a_d,
                           const int* __restrict__ src, const int* __restrict__ dst,
                           const float* __restrict__ mmax, float* __restrict__ ex,
                           float* __restrict__ denom, long total) {
    long i = (long)blockIdx.x * blockDim.x + threadIdx.x;
    if (i >= total) return;
    int e = (int)(i >> 2), h = (int)(i & 3);
    int d = dst[e];
    float v = leaky(a_s[(long)src[e] * 4 + h] + a_d[(long)d * 4 + h]);
    float xv = __expf(v - mmax[(long)d * 4 + h]);
    ex[i] = xv;
    atomicAdd(&denom[(long)d * 4 + h], xv);
    // warm L2 with the start of denom, which the scatter kernel reads
    __builtin_prefetch(denom, 0, 1);
}

// gout[dst,h,:] += hp[src,h,:] * alpha[e,h]
__global__ void k_gat_scatter(const float* __restrict__ hp, const float* __restrict__ ex,
                              const float* __restrict__ denom, const int* __restrict__ src,
                              const int* __restrict__ dst, float* __restrict__ gout,
                              long total) {
    long i = (long)blockIdx.x * blockDim.x + threadIdx.x;
    if (i >= total) return;
    int e = (int)(i >> 8);
    int t = (int)(i & 255);
    int h = t >> 6;
    int s = src[e], d = dst[e];
    float alpha = ex[(long)e * 4 + h] / (denom[(long)d * 4 + h] + 1e-16f);
    atomicAdd(&gout[(long)d * GPROJ + t], hp[(long)s * GPROJ + t] * alpha);
}

// h2 = relu(BN(mean_heads(gout) + b2))
__global__ void k_gat_epi(const float* __restrict__ gout, const float* __restrict__ b2,
                          const float* __restrict__ g2, const float* __restrict__ bt2,
                          float* __restrict__ h2, int total) {
    int i = blockIdx.x * blockDim.x + threadIdx.x;
    if (i >= total) return;
    int n = i >> 6, d = i & 63;
    float v = 0.0f;
#pragma unroll
    for (int h = 0; h < GHEADS; ++h) v += gout[(long)n * GPROJ + h * GHD + d];
    v = v * 0.25f + b2[d];
    v = v * (g2[d] * rsqrtf(1.0f + GEPS)) + bt2[d];
    h2[i] = fmaxf(v, 0.0f);
}

// --------------- fused skip + residual + classifier + log_softmax -----------
__global__ void k_final(const float* __restrict__ x, const float* __restrict__ h3,
                        const _Float16* __restrict__ Wskipp, const float* __restrict__ bskip,
                        const _Float16* __restrict__ Wc1p, const float* __restrict__ bc1,
                        const float* __restrict__ Wc2, const float* __restrict__ bc2,
                        float* __restrict__ out) {
    __shared__ float lds[16 * 32];
    int lane = threadIdx.x;
    int m0   = blockIdx.x * 16;
    int rowA = m0 + (lane & 15);

    // identity = x @ Wskip + bskip ; h = h3 + identity
    v16h Ax = load_a_row(x + (long)rowA * GIN, 0, GIN, lane, 1.0f);
    float hbuf[2][8];
#pragma unroll
    for (int t = 0; t < 2; ++t) {
        v8f acc = {};
        acc = WMMA_F16(Ax, load_frag(Wskipp, t, lane), acc);
        int n = t * 16 + (lane & 15);
#pragma unroll
        for (int r = 0; r < 8; ++r) {
            int m = m0 + r + ((lane >> 4) << 3);
            hbuf[t][r] = acc[r] + bskip[n] + h3[(long)m * GH3 + n];
        }
    }
    // transpose via LDS to rebuild A layout
#pragma unroll
    for (int t = 0; t < 2; ++t) {
        int n = t * 16 + (lane & 15);
#pragma unroll
        for (int r = 0; r < 8; ++r) lds[(r + ((lane >> 4) << 3)) * 32 + n] = hbuf[t][r];
    }
    __syncthreads();
    v16h Ah = load_a_row(lds + (lane & 15) * 32, 0, 32, lane, 1.0f);
    __syncthreads();

    // t = relu(h @ Wc1 + bc1)
    float tbuf[2][8];
#pragma unroll
    for (int t = 0; t < 2; ++t) {
        v8f acc = {};
        acc = WMMA_F16(Ah, load_frag(Wc1p, t, lane), acc);
        int n = t * 16 + (lane & 15);
#pragma unroll
        for (int r = 0; r < 8; ++r) tbuf[t][r] = fmaxf(acc[r] + bc1[n], 0.0f);
    }
#pragma unroll
    for (int t = 0; t < 2; ++t) {
        int n = t * 16 + (lane & 15);
#pragma unroll
        for (int r = 0; r < 8; ++r) lds[(r + ((lane >> 4) << 3)) * 32 + n] = tbuf[t][r];
    }
    __syncthreads();

    // logits (32 -> 2) + log_softmax; lane = (row, out-class)
    int r = lane & 15, o = lane >> 4;
    float logit = bc2[o];
#pragma unroll
    for (int k = 0; k < 32; ++k) logit += lds[r * 32 + k] * Wc2[k * 2 + o];
    float other = __shfl_xor(logit, 16);
    float mx  = fmaxf(logit, other);
    float lse = mx + __logf(__expf(logit - mx) + __expf(other - mx));
    out[(long)(m0 + r) * GOUT + o] = logit - lse;
}

// --------------------------------- driver -----------------------------------

static inline int nb(long total, int bs) { return (int)((total + bs - 1) / bs); }
static inline long roundup16(long v) { return (v + 15) & ~15L; }

extern "C" void kernel_launch(void* const* d_in, const int* in_sizes, int n_in,
                              void* d_out, int out_size, void* d_ws, size_t ws_size,
                              hipStream_t stream) {
    const float* x       = (const float*)d_in[0];
    const int*   ei      = (const int*)d_in[1];
    const float* Wl1     = (const float*)d_in[2];
    const float* Wr1     = (const float*)d_in[3];
    const float* b1      = (const float*)d_in[4];
    const float* g1      = (const float*)d_in[5];
    const float* bt1     = (const float*)d_in[6];
    const float* W2      = (const float*)d_in[7];
    const float* att_src = (const float*)d_in[8];
    const float* att_dst = (const float*)d_in[9];
    const float* b2      = (const float*)d_in[10];
    const float* g2      = (const float*)d_in[11];
    const float* bt2     = (const float*)d_in[12];
    const float* Wl3     = (const float*)d_in[13];
    const float* Wr3     = (const float*)d_in[14];
    const float* b3      = (const float*)d_in[15];
    const float* g3      = (const float*)d_in[16];
    const float* bt3     = (const float*)d_in[17];
    const float* Wskip   = (const float*)d_in[18];
    const float* bskip   = (const float*)d_in[19];
    const float* Wc1     = (const float*)d_in[20];
    const float* bc1     = (const float*)d_in[21];
    const float* Wc2     = (const float*)d_in[22];
    const float* bc2     = (const float*)d_in[23];

    const int* src = ei;
    const int* dst = ei + GE;
    float* out = (float*)d_out;

    const int MT = GN / 16;   // 3125 node tiles

    // ----- workspace layout (float units, 64B-aligned blocks) -----
    float* ws = (float*)d_ws;
    long off = 0;
    #define CARVE(name, nfloats) float* name = ws + off; off = roundup16(off + (nfloats))
    CARVE(deg,   GN);
    CARVE(agg1,  (long)GN * GIN);
    CARVE(hp,    (long)GN * GPROJ);
    CARVE(a_s,   (long)GN * GHEADS);
    CARVE(a_d,   (long)GN * GHEADS);
    CARVE(mmax,  (long)GN * GHEADS);
    CARVE(denom, (long)GN * GHEADS);
    CARVE(ex,    (long)GE * GHEADS);
    CARVE(gout,  (long)GN * GPROJ);
    CARVE(h2,    (long)GN * GHID);
    CARVE(agg3,  (long)GN * GHID);
    CARVE(h3,    (long)GN * GH3);
    CARVE(h1p_f, (long)MT * 2 * 32 * 16 / 2);   // f16 A-fragments of h1
    CARVE(wl1p_f,  4 * 512 / 2);
    CARVE(wr1p_f,  4 * 512 / 2);
    CARVE(w2p_f,  32 * 512 / 2);
    CARVE(wl3p_f,  4 * 512 / 2);
    CARVE(wr3p_f,  4 * 512 / 2);
    CARVE(wskp_f,  2 * 512 / 2);
    CARVE(wc1p_f,  2 * 512 / 2);
    #undef CARVE
    _Float16* h1p   = (_Float16*)h1p_f;
    _Float16* Wl1p  = (_Float16*)wl1p_f;
    _Float16* Wr1p  = (_Float16*)wr1p_f;
    _Float16* W2p   = (_Float16*)w2p_f;
    _Float16* Wl3p  = (_Float16*)wl3p_f;
    _Float16* Wr3p  = (_Float16*)wr3p_f;
    _Float16* Wskpp = (_Float16*)wskp_f;
    _Float16* Wc1p  = (_Float16*)wc1p_f;

    const int BS = 256;

    // ----- pre-pack weights into f16 B fragments (tiny, once per launch) ----
    k_packB<<<nb(4  * 32, BS), BS, 0, stream>>>(Wl1,   GIN,  GHID,  Wl1p,  4,  1);
    k_packB<<<nb(4  * 32, BS), BS, 0, stream>>>(Wr1,   GIN,  GHID,  Wr1p,  4,  1);
    k_packB<<<nb(32 * 32, BS), BS, 0, stream>>>(W2,    GHID, GPROJ, W2p,   32, 2);
    k_packB<<<nb(4  * 32, BS), BS, 0, stream>>>(Wl3,   GHID, GH3,   Wl3p,  4,  2);
    k_packB<<<nb(4  * 32, BS), BS, 0, stream>>>(Wr3,   GHID, GH3,   Wr3p,  4,  2);
    k_packB<<<nb(2  * 32, BS), BS, 0, stream>>>(Wskip, GIN,  GH3,   Wskpp, 2,  1);
    k_packB<<<nb(2  * 32, BS), BS, 0, stream>>>(Wc1,   GH3,  GH3,   Wc1p,  2,  1);

    // ----- init accumulators every call (graph replays don't re-zero) ------
    k_fill<<<nb(GN, BS), BS, 0, stream>>>(deg, GN, 0.0f);
    k_fill<<<nb((long)GN * GIN, BS), BS, 0, stream>>>(agg1, (long)GN * GIN, 0.0f);
    k_fill<<<nb((long)GN * GHEADS, BS), BS, 0, stream>>>(denom, (long)GN * GHEADS, 0.0f);
    k_fill<<<nb((long)GN * GPROJ, BS), BS, 0, stream>>>(gout, (long)GN * GPROJ, 0.0f);
    k_fill<<<nb((long)GN * GHID, BS), BS, 0, stream>>>(agg3, (long)GN * GHID, 0.0f);
    k_fill<<<nb((long)GN * GHEADS, BS), BS, 0, stream>>>(mmax, (long)GN * GHEADS, -3.0e38f);

    // degrees (+ L2 prefetch of x[src] rows for the next gather)
    k_degree<<<nb(GE, BS), BS, 0, stream>>>(dst, src, x, deg, GE);

    // SAGE1: aggregate then WMMA GEMM -> h1 stored as f16 A-fragments
    k_scatter_add<<<nb((long)GE * GIN, BS), BS, 0, stream>>>(x, src, dst, agg1,
                                                             (long)GE * GIN, GIN);
    k_sage<GIN, GHID, true><<<MT, 32, 0, stream>>>(agg1, x, deg, Wl1p, Wr1p,
                                                   b1, g1, bt1, nullptr, h1p);

    // GAT
    k_gatproj<<<dim3(MT, 4), 32, 0, stream>>>(h1p, W2p, hp);
    k_attdot<<<nb((long)GN * GHEADS, BS), BS, 0, stream>>>(hp, att_src, att_dst, a_s, a_d,
                                                           GN * GHEADS);
    k_edge_max<<<nb((long)GE * GHEADS, BS), BS, 0, stream>>>(a_s, a_d, src, dst, mmax,
                                                             (long)GE * GHEADS);
    k_edge_exp<<<nb((long)GE * GHEADS, BS), BS, 0, stream>>>(a_s, a_d, src, dst, mmax, ex,
                                                             denom, (long)GE * GHEADS);
    k_gat_scatter<<<nb((long)GE * GPROJ, BS), BS, 0, stream>>>(hp, ex, denom, src, dst, gout,
                                                               (long)GE * GPROJ);
    k_gat_epi<<<nb((long)GN * GHID, BS), BS, 0, stream>>>(gout, b2, g2, bt2, h2, GN * GHID);

    // SAGE3
    k_scatter_add<<<nb((long)GE * GHID, BS), BS, 0, stream>>>(h2, src, dst, agg3,
                                                              (long)GE * GHID, GHID);
    k_sage<GHID, GH3, false><<<MT, 32, 0, stream>>>(agg3, h2, deg, Wl3p, Wr3p,
                                                    b3, g3, bt3, h3, nullptr);

    // skip + residual + classifier + log_softmax
    k_final<<<MT, 32, 0, stream>>>(x, h3, Wskpp, bskip, Wc1p, bc1, Wc2, bc2, out);

    (void)in_sizes; (void)n_in; (void)out_size; (void)ws_size;
}
